// Self_Att_10118942949621
// MI455X (gfx1250) — matrix operand order repaired and checked by reading, not compiled, not run on hardware
//
#include <hip/hip_runtime.h>
#include <math.h>

typedef __attribute__((ext_vector_type(16))) _Float16 v16h;
typedef __attribute__((ext_vector_type(8)))  _Float16 v8h;
typedef __attribute__((ext_vector_type(4)))  _Float16 v4h;
typedef __attribute__((ext_vector_type(8)))  float    v8f;

#define H_     8
#define HD_    128
#define B_     1024
#define C_     1024
#define SCALE_ 0.08838834764831845f   // 128^-0.5
#define LOG2E_ 1.4426950408889634f
#define WPAD_  136                    // halves per LDS weight row (272B = 68 dwords -> +4-bank/row)

// ---------------- fp32 -> f16 conversion (vectorized x4) ----------------
__global__ void cvt_f32_f16(const float* __restrict__ s, _Float16* __restrict__ d, int n4)
{
    int i = blockIdx.x * blockDim.x + threadIdx.x;
    if (i < n4) {
        float4 f = ((const float4*)s)[i];
        v4h o;
        o[0] = (_Float16)f.x; o[1] = (_Float16)f.y;
        o[2] = (_Float16)f.z; o[3] = (_Float16)f.w;
        *(v4h*)(d + 4 * i) = o;
    }
}

// assemble a 16-half fragment from two contiguous 8-half (16B) chunks
static __device__ __forceinline__ v16h ldfrag(const _Float16* p0, const _Float16* p1)
{
    v8h a = *(const v8h*)p0;
    v8h b = *(const v8h*)p1;
    v16h r;
#pragma unroll
    for (int i = 0; i < 8; ++i) { r[i] = a[i]; r[i + 8] = b[i]; }
    return r;
}

// ---------------- kernel 1: WMMA QKV projection ----------------
// Block = 8 waves, ALL on the same head; each wave owns one 16-batch tile.
// Grid = 8 heads x 8 batch-groups = 64 blocks. Per-head weights are staged
// into LDS once per block in 3 phases (Q,K,V) via async global->LDS b128
// copies (ASYNCcnt), then WMMA B-fragments come from LDS at ds latency.
__global__ __launch_bounds__(256) void qkv_proj(const _Float16* __restrict__ xh,
                                                const _Float16* __restrict__ wqh,
                                                const _Float16* __restrict__ wkvh,
                                                float* __restrict__ qw,
                                                float* __restrict__ kw,
                                                float* __restrict__ vw)
{
    __shared__ _Float16 wlds[HD_ * WPAD_];   // 128 rows x 136 halves = 34816 B

    const int tid  = threadIdx.x;
    const int w    = tid >> 5;
    const int lane = tid & 31;
    const int h    = blockIdx.x & (H_ - 1);
    const int b0   = (((blockIdx.x >> 3) << 3) + w) << 4;   // wave's 16-batch tile
    const int hi   = lane >> 4;
    const int lo   = lane & 15;

    // ---- A fragments: X tile 16 batches x 32 d (ISA 16-bit A layout) ----
    // lane<16: row M=lo, K {0..7,16..23}; lane>=16: row M=lo, K {8..15,24..31}
    const _Float16* xp = xh + (size_t)(b0 + lo) * C_ + h * HD_;
    v16h afrag[4];
#pragma unroll
    for (int kc = 0; kc < 4; ++kc) {
        int d0 = kc * 32 + hi * 8;
        afrag[kc] = ldfrag(xp + d0, xp + d0 + 16);
    }

    const _Float16* wsrc[3];
    wsrc[0] = wqh + (size_t)h * HD_ * HD_;              // Q weights (128 rows)
    wsrc[1] = wkvh + (size_t)h * 2 * HD_ * HD_;         // K weights (Wkv rows 0..127)
    wsrc[2] = wsrc[1] + HD_ * HD_;                      // V weights (Wkv rows 128..255)
    float* obase[3];
    obase[0] = qw + ((size_t)h * B_ + b0) * HD_;
    obase[1] = kw + ((size_t)h * B_ + b0) * HD_;
    obase[2] = vw + ((size_t)h * B_ + b0) * HD_;

#pragma unroll
    for (int phase = 0; phase < 3; ++phase) {
        // ---- async-stage 128x128 f16 weights (32KB) into padded LDS ----
        // 2048 16-byte chunks, 8 per thread; ASYNCcnt tracks completion.
        const _Float16* ws = wsrc[phase];
#pragma unroll
        for (int j = 0; j < 8; ++j) {
            int ci  = j * 256 + tid;
            int row = ci >> 4;
            int cp  = ci & 15;
            unsigned    ldsb = (unsigned)(uintptr_t)&wlds[row * WPAD_ + cp * 8];
            const void* gsrc = (const char*)ws + row * 256 + cp * 16;
            asm volatile("global_load_async_to_lds_b128 %0, %1, off"
                         :: "v"(ldsb), "v"(gsrc)
                         : "memory");
        }
        asm volatile("s_wait_asynccnt 0x0" ::: "memory");
        __syncthreads();

        // ---- 8 N-chunks x 4 K-chunks of WMMA, B from LDS ----
        // B column N=e is weight row e; lane<16: col lo K {0..7,16..23}, lane>=16: col lo K {8..15,24..31}
#pragma unroll
        for (int nc = 0; nc < 8; ++nc) {
            const _Float16* wrow = &wlds[(nc * 16 + lo) * WPAD_ + hi * 8];
            v8f acc = {};
#pragma unroll
            for (int kc = 0; kc < 4; ++kc) {
                v16h bf = ldfrag(wrow + kc * 32, wrow + kc * 32 + 16);
                acc = __builtin_amdgcn_wmma_f32_16x16x32_f16(false, afrag[kc],
                                                             false, bf,
                                                             (short)0, acc,
                                                             false, false);
            }
            // D layout: lane<16 -> col lo, rows 0..7 ; lane>=16 -> col lo, rows 8..15
            const int col = nc * 16 + lo;
            float* orow = obase[phase] + (size_t)(hi * 8) * HD_ + col;
#pragma unroll
            for (int r = 0; r < 8; ++r)
                orow[(size_t)r * HD_] = acc[r];
        }
        __syncthreads();   // LDS reused by next phase
    }
}

// ---------------- kernel 2: softmax attention ----------------
// One wave per (b,h): 8192 waves = 1024 blocks x 8 waves.
// Separable logits L[d,e] = (q_d*scale)*k_e -> exact row max from kmax/kmin.
// Base-2 exponent, log2e prefolded: one v_exp + one fma per logit element.
__global__ __launch_bounds__(256) void attn_softmax(const float* __restrict__ qw,
                                                    const float* __restrict__ kw,
                                                    const float* __restrict__ vw,
                                                    float* __restrict__ out)
{
    __shared__ float kbuf[8][HD_];
    __shared__ float vbuf[8][HD_];

    const int w    = threadIdx.x >> 5;
    const int lane = threadIdx.x & 31;
    const int wave = blockIdx.x * 8 + w;
    const int h    = wave & (H_ - 1);
    const int b    = wave >> 3;

    const size_t roff = ((size_t)h * B_ + b) * HD_;
    float4 kk4 = ((const float4*)(kw + roff))[lane];
    float4 vv4 = ((const float4*)(vw + roff))[lane];
    ((float4*)kbuf[w])[lane] = kk4;
    ((float4*)vbuf[w])[lane] = vv4;

    // wave32 max/min reduction over k
    float km = fmaxf(fmaxf(kk4.x, kk4.y), fmaxf(kk4.z, kk4.w));
    float kn = fminf(fminf(kk4.x, kk4.y), fminf(kk4.z, kk4.w));
#pragma unroll
    for (int off = 16; off > 0; off >>= 1) {
        km = fmaxf(km, __shfl_xor(km, off, 32));
        kn = fminf(kn, __shfl_xor(kn, off, 32));
    }

    float4 qq = ((const float4*)(qw + roff))[lane];
    float a2[4] = { qq.x, qq.y, qq.z, qq.w };
    float m2[4], den[4], num[4];
#pragma unroll
    for (int i = 0; i < 4; ++i) {
        a2[i] *= SCALE_ * LOG2E_;                          // base-2 prescale
        m2[i]  = (a2[i] >= 0.f) ? a2[i] * km : a2[i] * kn; // exact row max (log2 domain)
        den[i] = 0.f;
        num[i] = 0.f;
    }

    __syncthreads();

    // 4 independent accumulator chains per lane, 8 exps per iteration
    for (int e = 0; e < HD_; e += 2) {
        float2 kk = *(const float2*)&kbuf[w][e];
        float2 vv = *(const float2*)&vbuf[w][e];
#pragma unroll
        for (int i = 0; i < 4; ++i) {
            float p0 = exp2f(fmaf(a2[i], kk.x, -m2[i]));
            float p1 = exp2f(fmaf(a2[i], kk.y, -m2[i]));
            den[i] += p0 + p1;
            num[i] = fmaf(p0, vv.x, fmaf(p1, vv.y, num[i]));
        }
    }

    float4 o = { num[0] / den[0], num[1] / den[1], num[2] / den[2], num[3] / den[3] };
    ((float4*)(out + (size_t)b * C_ + h * HD_))[lane] = o;
}

extern "C" void kernel_launch(void* const* d_in, const int* in_sizes, int n_in,
                              void* d_out, int out_size, void* d_ws, size_t ws_size,
                              hipStream_t stream)
{
    const float* x   = (const float*)d_in[0];   // (B, 1, C)
    const float* Wq  = (const float*)d_in[1];   // (H, HD, HD)
    const float* Wkv = (const float*)d_in[2];   // (H, 2HD, HD)
    float* out = (float*)d_out;                 // (B, 1, C) f32

    const int NX   = B_ * C_;             // 1048576
    const int NWQ  = H_ * HD_ * HD_;      // 131072
    const int NWKV = H_ * 2 * HD_ * HD_;  // 262144

    // workspace: f16 copies of x/Wq/Wkv, then fp32 Q/K/V laid out [h][b][e]
    _Float16* xh   = (_Float16*)d_ws;
    _Float16* wqh  = xh + NX;
    _Float16* wkvh = wqh + NWQ;
    float* qw = (float*)(wkvh + NWKV);
    float* kw = qw + (size_t)H_ * B_ * HD_;
    float* vw = kw + (size_t)H_ * B_ * HD_;

    cvt_f32_f16<<<(NX / 4 + 255) / 256, 256, 0, stream>>>(x, xh, NX / 4);
    cvt_f32_f16<<<(NWQ / 4 + 255) / 256, 256, 0, stream>>>(Wq, wqh, NWQ / 4);
    cvt_f32_f16<<<(NWKV / 4 + 255) / 256, 256, 0, stream>>>(Wkv, wkvh, NWKV / 4);

    // 512 projection waves (WMMA + async LDS staging), then 8192 attention waves
    qkv_proj<<<64, 256, 0, stream>>>(xh, wqh, wkvh, qw, kw, vw);
    attn_softmax<<<(B_ * H_) / 8, 256, 0, stream>>>(qw, kw, vw, out);
}
